// MixtralSparseMoeBlock_9929964388840
// MI455X (gfx1250) — compile-verified, matrix operand order checked by reading, not run on hardware
//
#include <hip/hip_runtime.h>
#include <hip/hip_bf16.h>
#include <math.h>

// ---------------- CDNA5 WMMA types ----------------
typedef __attribute__((ext_vector_type(16))) __bf16 v16bf;
typedef __attribute__((ext_vector_type(8)))  __bf16 v8bf;
typedef __attribute__((ext_vector_type(8)))  float  v8f;

#define T_TOK 4096
#define EMBED 1024
#define FFN   2048
#define NEXP  8

#define BM 128
#define BN 128
#define BK 32
#define LDT 40   // padded LDS row stride (bf16 elems): 80B, keeps 16B alignment

__device__ __forceinline__ unsigned short f2bf(float f) {
  unsigned int u = __float_as_uint(f);
  u += 0x7FFFu + ((u >> 16) & 1u);          // round-to-nearest-even
  return (unsigned short)(u >> 16);
}
__device__ __forceinline__ float bf2f(unsigned short h) {
  return __uint_as_float(((unsigned int)h) << 16);
}
// pack two floats as bf16 pair in one dword (lo in [15:0], hi in [31:16]).
// round-half-up (+0x8000) then one v_perm_b32 grabs both high halves.
__device__ __forceinline__ unsigned int pk2(float lo, float hi) {
  unsigned int ul = __float_as_uint(lo) + 0x8000u;
  unsigned int uh = __float_as_uint(hi) + 0x8000u;
  return __builtin_amdgcn_perm(uh, ul, 0x07060302u);  // {uh[3],uh[2],ul[3],ul[2]}
}

union AB16 { v16bf v; v8bf h[2]; };

// ISA 16-bit fragment map: lane&15 = row (A) / col (B, transposed-in-LDS),
// lanes<16 hold K = kb..kb+7 and kb+16..kb+23 with kb=0; lanes>=16 kb=8.
__device__ __forceinline__ v16bf ld_frag(const unsigned short* s, int row, int kb) {
  AB16 u;
  u.h[0] = *reinterpret_cast<const v8bf*>(s + row * LDT + kb);
  u.h[1] = *reinterpret_cast<const v8bf*>(s + row * LDT + kb + 16);
  return u.v;
}

__device__ __forceinline__ v8f wmma_bf16(v16bf a, v16bf b, v8f c) {
  return __builtin_amdgcn_wmma_f32_16x16x32_bf16(false, a, false, b, (short)0, c,
                                                 false, false);
}

// ---- A tile loader: 128x32 fp32 -> bf16 LDS, 2x ds_store_b128 per thread ----
__device__ __forceinline__ void load_a_f32(unsigned short* As, const float* A,
                                           int m0, int kk, int lda, int tid) {
  int row = tid >> 1;
  int c0  = (tid & 1) * 16;
  const float* g = A + (size_t)(m0 + row) * lda + kk + c0;
  __builtin_prefetch(g + BK, 0, 1);           // -> global_prefetch_b8
  float4 f0 = *reinterpret_cast<const float4*>(g);
  float4 f1 = *reinterpret_cast<const float4*>(g + 4);
  float4 f2 = *reinterpret_cast<const float4*>(g + 8);
  float4 f3 = *reinterpret_cast<const float4*>(g + 12);
  uint4 w0 = make_uint4(pk2(f0.x, f0.y), pk2(f0.z, f0.w),
                        pk2(f1.x, f1.y), pk2(f1.z, f1.w));
  uint4 w1 = make_uint4(pk2(f2.x, f2.y), pk2(f2.z, f2.w),
                        pk2(f3.x, f3.y), pk2(f3.z, f3.w));
  *reinterpret_cast<uint4*>(As + row * LDT + c0)     = w0;
  *reinterpret_cast<uint4*>(As + row * LDT + c0 + 8) = w1;
}

// ---- A tile loader, bf16 source (H): straight b128 copies ----
__device__ __forceinline__ void load_a_bf16(unsigned short* As, const unsigned short* A,
                                            int m0, int kk, int lda, int tid) {
  int row = tid >> 1;
  int c0  = (tid & 1) * 16;
  const unsigned short* g = A + (size_t)(m0 + row) * lda + kk + c0;
  __builtin_prefetch(g + BK, 0, 1);
  *reinterpret_cast<v8bf*>(As + row * LDT + c0) =
      *reinterpret_cast<const v8bf*>(g);
  *reinterpret_cast<v8bf*>(As + row * LDT + c0 + 8) =
      *reinterpret_cast<const v8bf*>(g + 8);
}

// ---- B tile [BK x BN] from row-major weights, stored TRANSPOSED: Bs[n][k].
// Each thread: 4 coalesced float4 loads (4 K-rows x 4 N-cols), register 4x4
// transpose, pack along K, 4x ds_store_b64.
__device__ __forceinline__ void load_b_f32(unsigned short* Bs, const float* B,
                                           int n0, int kk, int ldb, int tid) {
  int n4 = (tid & 31) * 4;      // 0..124 (contiguous across the wave)
  int k4 = (tid >> 5) * 4;      // 0..28  (wave id picks the K strip)
  const float* g = B + (size_t)(kk + k4) * ldb + n0 + n4;
  __builtin_prefetch(g + (size_t)BK * ldb, 0, 1);
  float4 r0 = *reinterpret_cast<const float4*>(g);
  float4 r1 = *reinterpret_cast<const float4*>(g + ldb);
  float4 r2 = *reinterpret_cast<const float4*>(g + 2 * (size_t)ldb);
  float4 r3 = *reinterpret_cast<const float4*>(g + 3 * (size_t)ldb);
  uint2 c0 = make_uint2(pk2(r0.x, r1.x), pk2(r2.x, r3.x));
  uint2 c1 = make_uint2(pk2(r0.y, r1.y), pk2(r2.y, r3.y));
  uint2 c2 = make_uint2(pk2(r0.z, r1.z), pk2(r2.z, r3.z));
  uint2 c3 = make_uint2(pk2(r0.w, r1.w), pk2(r2.w, r3.w));
  *reinterpret_cast<uint2*>(Bs + (n4 + 0) * LDT + k4) = c0;
  *reinterpret_cast<uint2*>(Bs + (n4 + 1) * LDT + k4) = c1;
  *reinterpret_cast<uint2*>(Bs + (n4 + 2) * LDT + k4) = c2;
  *reinterpret_cast<uint2*>(Bs + (n4 + 3) * LDT + k4) = c3;
}

// ---------------- kernel 0: zero output region ----------------
__global__ __launch_bounds__(256) void zero_out_kernel(float4* __restrict__ o) {
  o[(size_t)blockIdx.x * 256 + threadIdx.x] = make_float4(0.f, 0.f, 0.f, 0.f);
}

// ---------------- kernel 1: G = elu(x @ gw1 + gb1), bf16 ----------------
__global__ __launch_bounds__(256)
void gate_gemm(const float* __restrict__ X, const float* __restrict__ GW1,
               const float* __restrict__ GB1, unsigned short* __restrict__ G) {
  __shared__ alignas(16) unsigned short As[BM * LDT];
  __shared__ alignas(16) unsigned short Bs[BN * LDT];
  int tid = threadIdx.x, lane = tid & 31, wave = tid >> 5;
  int wm = wave & 3, wn = wave >> 2;
  int m0 = blockIdx.y * BM, n0 = blockIdx.x * BN;
  v8f acc[2][4] = {};
  int l15 = lane & 15, kb = (lane >> 4) << 3;

  for (int kk = 0; kk < EMBED; kk += BK) {
    load_a_f32(As, X, m0, kk, EMBED, tid);
    load_b_f32(Bs, GW1, n0, kk, 1024, tid);
    __syncthreads();
    v16bf af0 = ld_frag(As, 32 * wm + l15, kb);
    v16bf af1 = ld_frag(As, 32 * wm + 16 + l15, kb);
    #pragma unroll
    for (int sn = 0; sn < 4; ++sn) {
      v16bf bf = ld_frag(Bs, 64 * wn + 16 * sn + l15, kb);
      acc[0][sn] = wmma_bf16(af0, bf, acc[0][sn]);
      acc[1][sn] = wmma_bf16(af1, bf, acc[1][sn]);
    }
    __syncthreads();
  }
  float bias[4];
  #pragma unroll
  for (int sn = 0; sn < 4; ++sn) bias[sn] = GB1[n0 + 64 * wn + 16 * sn + l15];
  #pragma unroll
  for (int sm = 0; sm < 2; ++sm)
    #pragma unroll
    for (int i = 0; i < 8; ++i) {
      int row = m0 + 32 * wm + 16 * sm + ((lane >> 4) << 3) + i;
      #pragma unroll
      for (int sn = 0; sn < 4; ++sn) {
        int col = n0 + 64 * wn + 16 * sn + l15;
        float v = acc[sm][sn][i] + bias[sn];
        v = v > 0.f ? v : (expf(v) - 1.f);          // ELU
        G[(size_t)row * 1024 + col] = f2bf(v);
      }
    }
}

// ---------------- kernel 2: router logits = G @ gw2 + gb2 ----------------
__global__ __launch_bounds__(256)
void router_logits_kernel(const unsigned short* __restrict__ G,
                          const float* __restrict__ GW2,
                          const float* __restrict__ GB2,
                          float* __restrict__ L) {
  int idx = blockIdx.x * 256 + threadIdx.x;
  if (idx >= T_TOK * NEXP) return;
  int t = idx >> 3, e = idx & 7;
  const unsigned short* g = G + (size_t)t * 1024;
  float s = GB2[e];
  #pragma unroll 8
  for (int h = 0; h < 1024; ++h) s = fmaf(bf2f(g[h]), GW2[h * 8 + e], s);
  L[idx] = s;
}

// ---------------- kernel 3: softmax + top2 -> combine weights ----------------
__global__ __launch_bounds__(256)
void route_kernel(const float* __restrict__ L, float* __restrict__ C) {
  int t = blockIdx.x * 256 + threadIdx.x;
  if (t >= T_TOK) return;
  float l[NEXP], p[NEXP];
  float m = -3.4e38f;
  #pragma unroll
  for (int e = 0; e < NEXP; ++e) { l[e] = L[(size_t)t * NEXP + e]; m = fmaxf(m, l[e]); }
  float s = 0.f;
  #pragma unroll
  for (int e = 0; e < NEXP; ++e) { p[e] = expf(l[e] - m); s += p[e]; }
  #pragma unroll
  for (int e = 0; e < NEXP; ++e) p[e] /= s;   // softmax probs
  int i0 = 0;
  #pragma unroll
  for (int e = 1; e < NEXP; ++e) if (p[e] > p[i0]) i0 = e;
  int i1 = (i0 == 0) ? 1 : 0;
  #pragma unroll
  for (int e = 0; e < NEXP; ++e) if (e != i0 && p[e] > p[i1]) i1 = e;
  float ws = p[i0] + p[i1];
  #pragma unroll
  for (int e = 0; e < NEXP; ++e)
    C[(size_t)t * NEXP + e] = (e == i0) ? p[i0] / ws : (e == i1) ? p[i1] / ws : 0.f;
}

// -------- kernel 4: H = relu(x@w1) * (x@w3)  (dual-B, shared A loads) --------
__global__ __launch_bounds__(256)
void h_gemm(const float* __restrict__ X, const float* __restrict__ W1,
            const float* __restrict__ W3, unsigned short* __restrict__ H) {
  __shared__ alignas(16) unsigned short As [BM * LDT];
  __shared__ alignas(16) unsigned short Bs1[BN * LDT];
  __shared__ alignas(16) unsigned short Bs3[BN * LDT];
  int tid = threadIdx.x, lane = tid & 31, wave = tid >> 5;
  int wm = wave & 3, wn = wave >> 2;
  int m0 = blockIdx.y * BM, n0 = blockIdx.x * BN;
  v8f acc1[2][4] = {};
  v8f acc3[2][4] = {};
  int l15 = lane & 15, kb = (lane >> 4) << 3;

  for (int kk = 0; kk < EMBED; kk += BK) {
    load_a_f32(As, X, m0, kk, EMBED, tid);
    load_b_f32(Bs1, W1, n0, kk, FFN, tid);
    load_b_f32(Bs3, W3, n0, kk, FFN, tid);
    __syncthreads();
    v16bf af0 = ld_frag(As, 32 * wm + l15, kb);
    v16bf af1 = ld_frag(As, 32 * wm + 16 + l15, kb);
    #pragma unroll
    for (int sn = 0; sn < 4; ++sn) {
      int bc = 64 * wn + 16 * sn + l15;
      v16bf b1 = ld_frag(Bs1, bc, kb);
      acc1[0][sn] = wmma_bf16(af0, b1, acc1[0][sn]);
      acc1[1][sn] = wmma_bf16(af1, b1, acc1[1][sn]);
      v16bf b3 = ld_frag(Bs3, bc, kb);
      acc3[0][sn] = wmma_bf16(af0, b3, acc3[0][sn]);
      acc3[1][sn] = wmma_bf16(af1, b3, acc3[1][sn]);
    }
    __syncthreads();
  }
  #pragma unroll
  for (int sm = 0; sm < 2; ++sm)
    #pragma unroll
    for (int i = 0; i < 8; ++i) {
      int row = m0 + 32 * wm + 16 * sm + ((lane >> 4) << 3) + i;
      #pragma unroll
      for (int sn = 0; sn < 4; ++sn) {
        int col = n0 + 64 * wn + 16 * sn + l15;
        float v = fmaxf(acc1[sm][sn][i], 0.f) * acc3[sm][sn][i];
        H[(size_t)row * FFN + col] = f2bf(v);
      }
    }
}

// -------- kernel 5: out += combine[:,e] * (H @ w2) --------
__global__ __launch_bounds__(256)
void y_gemm(const unsigned short* __restrict__ H, const float* __restrict__ W2,
            const float* __restrict__ C, int e, float* __restrict__ OUT) {
  __shared__ alignas(16) unsigned short As[BM * LDT];
  __shared__ alignas(16) unsigned short Bs[BN * LDT];
  int tid = threadIdx.x, lane = tid & 31, wave = tid >> 5;
  int wm = wave & 3, wn = wave >> 2;
  int m0 = blockIdx.y * BM, n0 = blockIdx.x * BN;
  v8f acc[2][4] = {};
  int l15 = lane & 15, kb = (lane >> 4) << 3;

  for (int kk = 0; kk < FFN; kk += BK) {
    load_a_bf16(As, H, m0, kk, FFN, tid);
    load_b_f32(Bs, W2, n0, kk, EMBED, tid);
    __syncthreads();
    v16bf af0 = ld_frag(As, 32 * wm + l15, kb);
    v16bf af1 = ld_frag(As, 32 * wm + 16 + l15, kb);
    #pragma unroll
    for (int sn = 0; sn < 4; ++sn) {
      v16bf bf = ld_frag(Bs, 64 * wn + 16 * sn + l15, kb);
      acc[0][sn] = wmma_bf16(af0, bf, acc[0][sn]);
      acc[1][sn] = wmma_bf16(af1, bf, acc[1][sn]);
    }
    __syncthreads();
  }
  #pragma unroll
  for (int sm = 0; sm < 2; ++sm)
    #pragma unroll
    for (int i = 0; i < 8; ++i) {
      int row = m0 + 32 * wm + 16 * sm + ((lane >> 4) << 3) + i;
      float cw = C[(size_t)row * NEXP + e];
      if (cw != 0.f) {                        // top-2: ~75% of rows skipped
        #pragma unroll
        for (int sn = 0; sn < 4; ++sn) {
          int col = n0 + 64 * wn + 16 * sn + l15;
          size_t o = (size_t)row * EMBED + col;
          OUT[o] += cw * acc[sm][sn][i];      // expert kernels are stream-ordered
        }
      }
    }
}

// ---------------- host-side orchestration ----------------
extern "C" void kernel_launch(void* const* d_in, const int* in_sizes, int n_in,
                              void* d_out, int out_size, void* d_ws, size_t ws_size,
                              hipStream_t stream) {
  const float* x   = (const float*)d_in[0];   // [2,2048,1024]
  const float* gw1 = (const float*)d_in[1];   // [1024,1024]
  const float* gb1 = (const float*)d_in[2];   // [1024]
  const float* gw2 = (const float*)d_in[3];   // [1024,8]
  const float* gb2 = (const float*)d_in[4];   // [8]
  const float* w1  = (const float*)d_in[5];   // [8,1024,2048]
  const float* w3  = (const float*)d_in[6];   // [8,1024,2048]
  const float* w2  = (const float*)d_in[7];   // [8,2048,1024]

  float* out    = (float*)d_out;                       // [4096,1024]
  float* logits = out + (size_t)T_TOK * EMBED;         // [4096,8] (tuple output #2)

  // workspace: G (bf16, 8MB) lives in first half, overlaid later by H (bf16, 16MB)
  unsigned short* G  = (unsigned short*)d_ws;
  unsigned short* Hb = (unsigned short*)d_ws;
  float* combine = (float*)((char*)d_ws + (size_t)T_TOK * FFN * sizeof(unsigned short));

  zero_out_kernel<<<(T_TOK * EMBED) / (256 * 4), 256, 0, stream>>>((float4*)out);

  gate_gemm<<<dim3(1024 / BN, T_TOK / BM), 256, 0, stream>>>(x, gw1, gb1, G);
  router_logits_kernel<<<(T_TOK * NEXP + 255) / 256, 256, 0, stream>>>(G, gw2, gb2, logits);
  route_kernel<<<(T_TOK + 255) / 256, 256, 0, stream>>>(logits, combine);

  for (int e = 0; e < NEXP; ++e) {
    const float* w1e = w1 + (size_t)e * EMBED * FFN;
    const float* w3e = w3 + (size_t)e * EMBED * FFN;
    const float* w2e = w2 + (size_t)e * FFN * EMBED;
    h_gemm<<<dim3(FFN / BN, T_TOK / BM), 256, 0, stream>>>(x, w1e, w3e, Hb);
    y_gemm<<<dim3(EMBED / BN, T_TOK / BM), 256, 0, stream>>>(Hb, w2e, combine, e, out);
  }
}
